// Model_21672404975748
// MI455X (gfx1250) — matrix operand order repaired and checked by reading, not compiled, not run on hardware
//
#include <hip/hip_runtime.h>

// ---------------- CDNA5 WMMA types ----------------
typedef __attribute__((ext_vector_type(16))) __bf16 v16bf;
typedef __attribute__((ext_vector_type(8)))  float  v8f;

#define B_ROWS   4096
#define D_IN     2048
#define DIM      1024
#define N_EST    1000
#define NPAD     1024     // padded class dimension (rows 1000..1023 of C are zero)
#define TEMP_INV 10.0f    // 1/0.1
#define EPS_INV  20.0f    // 1/0.05

// GEMM tiling: block = 256 threads = 8 waves; block tile 128x128, K stage 64.
#define BM  128
#define BN  128
#define BK  64
#define LDK 72            // padded LDS row stride (elements); 144B, 16B aligned

// ---------------- bf16 helpers (bit-exact RNE, no reliance on cast lowering) ----
static __device__ __forceinline__ __bf16 f2bf(float f) {
  unsigned u = __float_as_uint(f);
  unsigned r = (u + 0x7fffu + ((u >> 16) & 1u)) >> 16;
  unsigned short s = (unsigned short)r;
  __bf16 h; __builtin_memcpy(&h, &s, 2); return h;
}
static __device__ __forceinline__ float bf2f(__bf16 h) {
  unsigned short s; __builtin_memcpy(&s, &h, 2);
  return __uint_as_float(((unsigned)s) << 16);
}

// ------------- CDNA5 async global->LDS copy (ASYNCcnt path, ISA 15.18.3) -------
// GV addressing: 64-bit global address in VGPR pair, LDS byte address in VDST.
// Low 32 bits of a generic pointer to LDS are the DS-space offset (ISA 10.2).
static __device__ __forceinline__ void async_copy_b128(void* lds_dst, const void* gsrc) {
  asm volatile("global_load_async_to_lds_b128 %0, %1, off"
               :: "v"((unsigned)(uintptr_t)lds_dst),
                  "v"((unsigned long long)(uintptr_t)gsrc)
               : "memory");
}
static __device__ __forceinline__ void wait_asynccnt0() {
  asm volatile("s_wait_asynccnt 0" ::: "memory");
}

// ---------------- block reductions (blockDim.x == 256) ----------------
static __device__ __forceinline__ float blk_sum(float v, float* sm) {
  int t = threadIdx.x;
  sm[t] = v; __syncthreads();
  for (int s = 128; s > 0; s >>= 1) { if (t < s) sm[t] += sm[t + s]; __syncthreads(); }
  float r = sm[0]; __syncthreads();
  return r;
}
static __device__ __forceinline__ float blk_max(float v, float* sm) {
  int t = threadIdx.x;
  sm[t] = v; __syncthreads();
  for (int s = 128; s > 0; s >>= 1) { if (t < s) sm[t] = fmaxf(sm[t], sm[t + s]); __syncthreads(); }
  float r = sm[0]; __syncthreads();
  return r;
}

// ---------------- WMMA fragment loads, contiguous k order (k = 16*(l>>4) + s) for A and B ----------------
union FragU { v16bf v; uint4 q[2]; };

// A 16x32 bf16: lane l -> row (l&15); K {kh..kh+15}, kh=(l>>4)*16
static __device__ __forceinline__ v16bf frag_load_a(const __bf16* p) {
  FragU f;
  f.q[0] = *(const uint4*)(p);
  f.q[1] = *(const uint4*)(p + 8);
  return f.v;
}
// B 32x16 bf16 (from B^T row-major): lane l -> col (l&15); K {koff..koff+15}, koff=(l>>4)*16
static __device__ __forceinline__ v16bf frag_load_b(const __bf16* p) {
  FragU f;
  f.q[0] = *(const uint4*)(p);
  f.q[1] = *(const uint4*)(p + 8);
  return f.v;
}

// 8-wave compute core: wave tile 32(M) x 64(N); 16 wmma per K stage of 64
static __device__ __forceinline__ void wmma_tile(const __bf16* sA, const __bf16* sB,
                                                 v8f acc[2][4]) {
  const int lane = threadIdx.x & 31;
  const int wave = threadIdx.x >> 5;
  const int wm = (wave & 3) * 32;     // 4 M slices
  const int wn = (wave >> 2) * 64;    // 2 N slices
  const int r  = lane & 15;
  const int kha = (lane >> 4) * 16;   // A half select
  const int khb = (lane >> 4) * 16;   // B half select
#pragma unroll
  for (int kk = 0; kk < BK; kk += 32) {
    v16bf a[2], b[4];
#pragma unroll
    for (int i = 0; i < 2; i++)
      a[i] = frag_load_a(sA + (wm + i * 16 + r) * LDK + kk + kha);
#pragma unroll
    for (int j = 0; j < 4; j++)
      b[j] = frag_load_b(sB + (wn + j * 16 + r) * LDK + kk + khb);
#pragma unroll
    for (int i = 0; i < 2; i++)
#pragma unroll
      for (int j = 0; j < 4; j++)
        acc[i][j] = __builtin_amdgcn_wmma_f32_16x16x32_bf16(
            false, a[i], false, b[j], (short)0, acc[i][j], false, false);
  }
}

// ---------------- kernel 1: W_b [2048,1024] f32 -> W_b^T [1024,2048] bf16 -------
__global__ __launch_bounds__(256) void k_wbT(const float* __restrict__ Wb,
                                             __bf16* __restrict__ WbT) {
  __shared__ float tile[32][33];
  const int n0 = blockIdx.x * 32, k0 = blockIdx.y * 32;
  const int tx = threadIdx.x & 31, ty = threadIdx.x >> 5;  // 32 x 8
  for (int j = 0; j < 32; j += 8)
    tile[ty + j][tx] = Wb[(size_t)(k0 + ty + j) * DIM + (n0 + tx)];
  __syncthreads();
  for (int j = 0; j < 32; j += 8)
    WbT[(size_t)(n0 + ty + j) * D_IN + (k0 + tx)] = f2bf(tile[tx][ty + j]);
}

// ---------------- kernel 2: normalized prototypes -> bf16 [1024,1024] ----------
__global__ __launch_bounds__(256) void k_protonorm(const float* __restrict__ Wp,
                                                   __bf16* __restrict__ C) {
  __shared__ float sm[256];
  const int r = blockIdx.x;
  if (r >= N_EST) {  // zero padding rows -> zero logits in padded columns
    for (int c = threadIdx.x; c < DIM; c += 256) C[(size_t)r * DIM + c] = f2bf(0.0f);
    return;
  }
  const float* row = Wp + (size_t)r * DIM;
  float v[4], ss = 0.f;
#pragma unroll
  for (int j = 0; j < 4; j++) { v[j] = row[threadIdx.x + j * 256]; ss += v[j] * v[j]; }
  ss = blk_sum(ss, sm);
  const float sc = 1.0f / fmaxf(sqrtf(ss), 1e-12f);
#pragma unroll
  for (int j = 0; j < 4; j++) C[(size_t)r * DIM + threadIdx.x + j * 256] = f2bf(v[j] * sc);
}

// -------- kernel 3: backbone GEMM, fused input-mix + f32->bf16 A staging -------
// Double-buffered LDS; B tile via async global->LDS; A tile f32->regs->bf16 LDS.
// variant z: 0:x1  1:x2  2:(x1+x2[idx])/2  3:(x2+x1[idx])/2 ; Z = A @ Wb (bf16 out)
__global__ __launch_bounds__(256) void k_gemm_backbone(
    const float* __restrict__ x1, const float* __restrict__ x2,
    const int* __restrict__ idx, const __bf16* __restrict__ BT,
    __bf16* __restrict__ Zall) {
  __shared__ __bf16 sA[2][BM * LDK];
  __shared__ __bf16 sB[2][BN * LDK];
  __shared__ int sIdx[BM];
  const int v = blockIdx.z;
  const int nBase = blockIdx.x * BN;
  const int rowBase = blockIdx.y * BM;
  const bool mix = (v >= 2);
  const float* P = (v == 0 || v == 2) ? x1 : x2;
  const float* S = (v == 2) ? x2 : x1;
  __bf16* Z = Zall + (size_t)v * B_ROWS * DIM;

  if (threadIdx.x < BM) sIdx[threadIdx.x] = idx[rowBase + threadIdx.x];
  __syncthreads();

  v8f zero = {};
  v8f acc[2][4];
#pragma unroll
  for (int i = 0; i < 2; i++)
#pragma unroll
    for (int j = 0; j < 4; j++) acc[i][j] = zero;

  float4 pReg[8], sReg[8];

  // issue global f32 loads for the A tile of stage with K offset k0 (held in regs)
  auto issueA = [&](int k0) {
#pragma unroll
    for (int i = 0; i < 8; i++) {
      const int c = threadIdx.x + i * 256;
      const int rr = c >> 4, cc = (c & 15) * 4;
      pReg[i] = *(const float4*)(P + (size_t)(rowBase + rr) * D_IN + k0 + cc);
      if (mix) sReg[i] = *(const float4*)(S + (size_t)sIdx[rr] * D_IN + k0 + cc);
    }
  };
  // mix + convert regs -> bf16 and store to LDS buffer
  auto storeA = [&](int buf) {
#pragma unroll
    for (int i = 0; i < 8; i++) {
      const int c = threadIdx.x + i * 256;
      const int rr = c >> 4, cc = (c & 15) * 4;
      float4 a4 = pReg[i];
      if (mix) {
        a4.x = 0.5f * (a4.x + sReg[i].x); a4.y = 0.5f * (a4.y + sReg[i].y);
        a4.z = 0.5f * (a4.z + sReg[i].z); a4.w = 0.5f * (a4.w + sReg[i].w);
      }
      union { __bf16 h[4]; uint2 u; } pk;
      pk.h[0] = f2bf(a4.x); pk.h[1] = f2bf(a4.y); pk.h[2] = f2bf(a4.z); pk.h[3] = f2bf(a4.w);
      *(uint2*)(&sA[buf][rr * LDK + cc]) = pk.u;
    }
  };
  // B tile: pure bf16 copy, async global->LDS (no VGPR round trip)
  auto issueB = [&](int k0, int buf) {
#pragma unroll
    for (int i = 0; i < 4; i++) {
      const int c = threadIdx.x + i * 256;
      const int rr = c >> 3, cc = (c & 7) * 8;
      async_copy_b128(&sB[buf][rr * LDK + cc],
                      BT + (size_t)(nBase + rr) * D_IN + k0 + cc);
    }
  };

  const int NST = D_IN / BK;  // 32 stages
  // prologue: stage 0
  issueA(0);
  issueB(0, 0);
  storeA(0);
  for (int s = 0; s < NST; s++) {
    wait_asynccnt0();     // own async copies for stage s complete
    __syncthreads();      // + own ds_stores (dscnt) ; stage s visible to all waves
    if (s + 1 < NST) {
      issueA((s + 1) * BK);          // f32 loads in flight across the WMMA block
      issueB((s + 1) * BK, (s + 1) & 1);
    }
    wmma_tile(sA[s & 1], sB[s & 1], acc);
    if (s + 1 < NST) storeA((s + 1) & 1);  // after compute's ds_loads (DS in-order)
  }

  // store bf16 (unnormalized Z); f32 C/D layout: VGPR e -> M = (lane>=16?8:0)+e, N=lane&15
  const int lane = threadIdx.x & 31;
  const int wave = threadIdx.x >> 5;
  const int wm = (wave & 3) * 32, wn = (wave >> 2) * 64;
  const int hh = lane >> 4, nn = lane & 15;
#pragma unroll
  for (int i = 0; i < 2; i++)
#pragma unroll
    for (int j = 0; j < 4; j++)
#pragma unroll
      for (int e = 0; e < 8; e++) {
        const int m = rowBase + wm + i * 16 + hh * 8 + e;
        const int n = nBase + wn + j * 16 + nn;
        Z[(size_t)m * DIM + n] = f2bf(acc[i][j][e]);
      }
}

// ---------------- kernel 4: row L2-normalize Z in place (bf16) ----------------
__global__ __launch_bounds__(256) void k_rownorm(__bf16* __restrict__ Zall) {
  __shared__ float sm[256];
  const int r = blockIdx.x, v = blockIdx.y;
  __bf16* row = Zall + (size_t)v * B_ROWS * DIM + (size_t)r * DIM;
  float vals[4], ss = 0.f;
#pragma unroll
  for (int j = 0; j < 4; j++) { vals[j] = bf2f(row[threadIdx.x + j * 256]); ss += vals[j] * vals[j]; }
  ss = blk_sum(ss, sm);
  const float sc = 1.0f / fmaxf(sqrtf(ss), 1e-12f);
#pragma unroll
  for (int j = 0; j < 4; j++) row[threadIdx.x + j * 256] = f2bf(vals[j] * sc);
}

// ------- kernel 5: logits GEMM L = Z @ C^T (f32 out), fully async staging ------
__global__ __launch_bounds__(256) void k_gemm_logits(
    const __bf16* __restrict__ Zall, const __bf16* __restrict__ C,
    float* __restrict__ Lall) {
  __shared__ __bf16 sA[2][BM * LDK];
  __shared__ __bf16 sB[2][BN * LDK];
  const int v = blockIdx.z;
  const int nBase = blockIdx.x * BN;
  const int rowBase = blockIdx.y * BM;
  const __bf16* A = Zall + (size_t)v * B_ROWS * DIM;
  float* L = Lall + (size_t)v * B_ROWS * NPAD;

  v8f zero = {};
  v8f acc[2][4];
#pragma unroll
  for (int i = 0; i < 2; i++)
#pragma unroll
    for (int j = 0; j < 4; j++) acc[i][j] = zero;

  auto issueAB = [&](int k0, int buf) {
#pragma unroll
    for (int i = 0; i < 4; i++) {
      const int c = threadIdx.x + i * 256;
      const int rr = c >> 3, cc = (c & 7) * 8;
      async_copy_b128(&sA[buf][rr * LDK + cc],
                      A + (size_t)(rowBase + rr) * DIM + k0 + cc);
      async_copy_b128(&sB[buf][rr * LDK + cc],
                      C + (size_t)(nBase + rr) * DIM + k0 + cc);
    }
  };

  const int NST = DIM / BK;  // 16 stages
  issueAB(0, 0);
  for (int s = 0; s < NST; s++) {
    wait_asynccnt0();
    __syncthreads();
    if (s + 1 < NST) issueAB((s + 1) * BK, (s + 1) & 1);
    wmma_tile(sA[s & 1], sB[s & 1], acc);
  }

  const int lane = threadIdx.x & 31;
  const int wave = threadIdx.x >> 5;
  const int wm = (wave & 3) * 32, wn = (wave >> 2) * 64;
  const int hh = lane >> 4, nn = lane & 15;
#pragma unroll
  for (int i = 0; i < 2; i++)
#pragma unroll
    for (int j = 0; j < 4; j++)
#pragma unroll
      for (int e = 0; e < 8; e++) {
        const int m = rowBase + wm + i * 16 + hh * 8 + e;
        const int n = nBase + wn + j * 16 + nn;
        L[(size_t)m * NPAD + n] = acc[i][j][e];
      }
}

// --------- kernel 6: q = exp(l/EPS) row-normalized (stable), pad cols = 0 -------
__global__ __launch_bounds__(256) void k_expnorm(const float* __restrict__ Lall,
                                                 float* __restrict__ Qall) {
  __shared__ float sm[256];
  const int r = blockIdx.x, m = blockIdx.y;
  const float* l = Lall + (size_t)m * B_ROWS * NPAD + (size_t)r * NPAD;
  float* q = Qall + (size_t)m * B_ROWS * NPAD + (size_t)r * NPAD;
  float v[4], mx = -3.4e38f;
#pragma unroll
  for (int j = 0; j < 4; j++) {
    const int c = threadIdx.x + j * 256;
    v[j] = (c < N_EST) ? l[c] : -3.4e38f;
    mx = fmaxf(mx, v[j]);
  }
  mx = blk_max(mx, sm);
  float e[4], s = 0.f;
#pragma unroll
  for (int j = 0; j < 4; j++) {
    const int c = threadIdx.x + j * 256;
    e[j] = (c < N_EST) ? __expf((v[j] - mx) * EPS_INV) : 0.f;
    s += e[j];
  }
  s = blk_sum(s, sm);
  const float inv = 1.0f / s;
#pragma unroll
  for (int j = 0; j < 4; j++) q[threadIdx.x + j * 256] = e[j] * inv;
}

// ---------------- sinkhorn: zero, colsum, fused col-div + row-norm -------------
__global__ void k_zero(float* __restrict__ p, int n) {
  const int i = blockIdx.x * blockDim.x + threadIdx.x;
  if (i < n) p[i] = 0.f;
}
__global__ __launch_bounds__(256) void k_colsum(const float* __restrict__ Qall,
                                                float* __restrict__ cs) {
  const int rb = blockIdx.x * 32, m = blockIdx.y;
  const float* Q = Qall + (size_t)m * B_ROWS * NPAD;
  float s[4] = {0.f, 0.f, 0.f, 0.f};
  for (int rr = 0; rr < 32; rr++) {
    const float* row = Q + (size_t)(rb + rr) * NPAD;
#pragma unroll
    for (int j = 0; j < 4; j++) {
      const int c = threadIdx.x + j * 256;
      if (c < N_EST) s[j] += row[c];
    }
  }
#pragma unroll
  for (int j = 0; j < 4; j++) {
    const int c = threadIdx.x + j * 256;
    if (c < N_EST) atomicAdd(&cs[m * NPAD + c], s[j]);
  }
}
__global__ __launch_bounds__(256) void k_colrow(float* __restrict__ Qall,
                                                const float* __restrict__ cs) {
  __shared__ float sm[256];
  const int r = blockIdx.x, m = blockIdx.y;
  float* q = Qall + (size_t)m * B_ROWS * NPAD + (size_t)r * NPAD;
  const float* col = cs + m * NPAD;
  float v[4], rs = 0.f;
#pragma unroll
  for (int j = 0; j < 4; j++) {
    const int c = threadIdx.x + j * 256;
    v[j] = (c < N_EST) ? q[c] / col[c] : 0.f;
    rs += v[j];
  }
  rs = blk_sum(rs, sm);
  const float inv = 1.0f / rs;
#pragma unroll
  for (int j = 0; j < 4; j++) {
    const int c = threadIdx.x + j * 256;
    if (c < N_EST) q[c] = v[j] * inv;
  }
}

// ---------------- labeled rows -> one-hot targets ----------------
__global__ __launch_bounds__(256) void k_label(float* __restrict__ Qall,
                                               const int* __restrict__ t) {
  const int r = blockIdx.x, m = blockIdx.y;
  const int tv = t[r];
  if (tv < 0) return;
  float* q = Qall + (size_t)m * B_ROWS * NPAD + (size_t)r * NPAD;
  for (int c = threadIdx.x; c < NPAD; c += 256) q[c] = (c == tv) ? 1.0f : 0.0f;
}

// ---------------- per-row loss: Σ_i [LSE(l_i/T)] - (1/T) Σ q·l ----------------
__global__ __launch_bounds__(256) void k_loss(const float* __restrict__ Lall,
                                              const float* __restrict__ Qall,
                                              const int* __restrict__ idx,
                                              float* __restrict__ partial) {
  __shared__ float sm[256];
  const int b = blockIdx.x;
  const int p = idx[b];
  const size_t S = (size_t)B_ROWS * NPAD;
  const float* l1  = Lall + 0 * S + (size_t)b * NPAD;
  const float* l2  = Lall + 1 * S + (size_t)b * NPAD;
  const float* l1m = Lall + 2 * S + (size_t)b * NPAD;
  const float* l2m = Lall + 3 * S + (size_t)b * NPAD;
  const float* q1b = Qall + 0 * S + (size_t)b * NPAD;
  const float* q2b = Qall + 1 * S + (size_t)b * NPAD;
  const float* q1p = Qall + 0 * S + (size_t)p * NPAD;
  const float* q2p = Qall + 1 * S + (size_t)p * NPAD;
  float m1 = -3.4e38f, m2 = m1, m3 = m1, m4 = m1;
#pragma unroll
  for (int j = 0; j < 4; j++) {
    const int c = threadIdx.x + j * 256;
    if (c < N_EST) {
      m1 = fmaxf(m1, l1[c]);  m2 = fmaxf(m2, l2[c]);
      m3 = fmaxf(m3, l1m[c]); m4 = fmaxf(m4, l2m[c]);
    }
  }
  m1 = blk_max(m1, sm); m2 = blk_max(m2, sm);
  m3 = blk_max(m3, sm); m4 = blk_max(m4, sm);
  float s1 = 0, s2 = 0, s3 = 0, s4 = 0, d = 0;
#pragma unroll
  for (int j = 0; j < 4; j++) {
    const int c = threadIdx.x + j * 256;
    if (c < N_EST) {
      const float a1 = l1[c], a2 = l2[c], a3 = l1m[c], a4 = l2m[c];
      s1 += __expf((a1 - m1) * TEMP_INV);
      s2 += __expf((a2 - m2) * TEMP_INV);
      s3 += __expf((a3 - m3) * TEMP_INV);
      s4 += __expf((a4 - m4) * TEMP_INV);
      d += q2b[c] * a1 + q1b[c] * a2
         + 0.5f * (q2b[c] + q1p[c]) * a3
         + 0.5f * (q1b[c] + q2p[c]) * a4;
    }
  }
  s1 = blk_sum(s1, sm); s2 = blk_sum(s2, sm);
  s3 = blk_sum(s3, sm); s4 = blk_sum(s4, sm);
  d = blk_sum(d, sm);
  if (threadIdx.x == 0) {
    partial[b] = (m1 + m2 + m3 + m4) * TEMP_INV
               + logf(s1) + logf(s2) + logf(s3) + logf(s4)
               - d * TEMP_INV;
  }
}

__global__ __launch_bounds__(256) void k_final(const float* __restrict__ partial,
                                               float* __restrict__ out) {
  __shared__ float sm[256];
  float s = 0.f;
  for (int i = threadIdx.x; i < B_ROWS; i += 256) s += partial[i];
  s = blk_sum(s, sm);
  if (threadIdx.x == 0) out[0] = s * (0.5f / (float)B_ROWS);
}

// ---------------- host launcher ----------------
extern "C" void kernel_launch(void* const* d_in, const int* in_sizes, int n_in,
                              void* d_out, int out_size, void* d_ws, size_t ws_size,
                              hipStream_t stream) {
  const float* x1  = (const float*)d_in[0];
  const float* x2  = (const float*)d_in[1];
  const int*   t   = (const int*)d_in[2];
  const int*   idx = (const int*)d_in[3];
  const float* Wb  = (const float*)d_in[4];
  const float* Wp  = (const float*)d_in[5];
  float* out = (float*)d_out;

  char* ws = (char*)d_ws;
  size_t off = 0;
  auto alloc = [&](size_t bytes) -> char* {
    char* p = ws + off;
    off = (off + bytes + 255) & ~(size_t)255;
    return p;
  };
  __bf16* WbT = (__bf16*)alloc((size_t)DIM * D_IN * 2);          //   4 MB
  __bf16* C   = (__bf16*)alloc((size_t)NPAD * DIM * 2);          //   2 MB
  __bf16* Z   = (__bf16*)alloc((size_t)4 * B_ROWS * DIM * 2);    //  32 MB
  float*  L   = (float*)alloc((size_t)4 * B_ROWS * NPAD * 4);    //  64 MB
  float*  Q   = (float*)alloc((size_t)2 * B_ROWS * NPAD * 4);    //  32 MB
  float*  cs  = (float*)alloc((size_t)2 * NPAD * 4);
  float*  prt = (float*)alloc((size_t)B_ROWS * 4);

  k_wbT<<<dim3(DIM / 32, D_IN / 32), dim3(256), 0, stream>>>(Wb, WbT);
  k_protonorm<<<dim3(NPAD), dim3(256), 0, stream>>>(Wp, C);
  k_gemm_backbone<<<dim3(DIM / BN, B_ROWS / BM, 4), dim3(256), 0, stream>>>(x1, x2, idx, WbT, Z);
  k_rownorm<<<dim3(B_ROWS, 4), dim3(256), 0, stream>>>(Z);
  k_gemm_logits<<<dim3(NPAD / BN, B_ROWS / BM, 4), dim3(256), 0, stream>>>(Z, C, L);
  k_expnorm<<<dim3(B_ROWS, 2), dim3(256), 0, stream>>>(L, Q);
  for (int it = 0; it < 3; it++) {
    k_zero<<<dim3((2 * NPAD + 255) / 256), dim3(256), 0, stream>>>(cs, 2 * NPAD);
    k_colsum<<<dim3(B_ROWS / 32, 2), dim3(256), 0, stream>>>(Q, cs);
    k_colrow<<<dim3(B_ROWS, 2), dim3(256), 0, stream>>>(Q, cs);
  }
  k_label<<<dim3(B_ROWS, 2), dim3(256), 0, stream>>>(Q, t);
  k_loss<<<dim3(B_ROWS), dim3(256), 0, stream>>>(L, Q, idx, prt);
  k_final<<<dim3(1), dim3(256), 0, stream>>>(prt, out);

  (void)in_sizes; (void)n_in; (void)out_size; (void)ws_size;
}
